// ODERNN_10608569221653
// MI455X (gfx1250) — compile-verified, matrix operand order checked by reading, not run on hardware
//
#include <hip/hip_runtime.h>
#include <hip/hip_bf16.h>

typedef _Float16 h16 __attribute__((ext_vector_type(16)));
typedef float    f8  __attribute__((ext_vector_type(8)));

#define B_SZ 1024
#define T_SZ 200
#define D_SZ 64
#define L_SZ 256
#define H_SZ 256
#define NSUB 4
#define BM   16      // batch rows per workgroup
#define ASTR 288     // padded K-stride (halves) for activation buffers (257 -> 288)
#define K1P  288     // padded K for layer-1 weights

// ws layout (in _Float16 elements, all offsets 32B-aligned)
#define WS_WIH 0                    // 768*64   = 49152
#define WS_WHH 49152                // 768*256  = 196608
#define WS_W1P 245760               // 256*288  = 73728 (zero padded)
#define WS_W2  319488               // 256*256  = 65536
#define WS_W3  385024               // 256*256  = 65536

// dynamic LDS layout (bytes)
#define SM_H     0        // 16*256*4  = 16384
#define SM_A     16384    // 16*288*2  =  9216
#define SM_B     25600    // 16*288*2  =  9216
#define SM_BIH   34816    // 3*256*4   =  3072
#define SM_BHH   37888    //            3072
#define SM_B1    40960    //            1024
#define SM_B2    41984    //            1024
#define SM_B3    43008    //            1024
#define SM_MASK  44032    // 16*4 -> pad to 64
#define SM_W2    44096    // 256*256*2 = 131072
#define SM_W3    175168   // 256*256*2 = 131072
#define SM_TOTAL 306240

__device__ __forceinline__ float sigm(float v) {
    return __builtin_amdgcn_rcpf(1.0f + __expf(-v));
}
__device__ __forceinline__ float tanh_fast(float v) {
    // tanh(v) = 1 - 2/(exp(2v)+1)
    return 1.0f - 2.0f * __builtin_amdgcn_rcpf(1.0f + __expf(2.0f * v));
}

template<int NK>
__device__ __forceinline__ void mma2(const _Float16* A,
                                     const _Float16* B0, const _Float16* B1,
                                     f8& c0, f8& c1) {
#pragma unroll
    for (int k = 0; k < NK; ++k) {
        h16 a  = *(const h16*)(A  + k * 32);
        h16 b0 = *(const h16*)(B0 + k * 32);
        h16 b1 = *(const h16*)(B1 + k * 32);
        c0 = __builtin_amdgcn_wmma_f32_16x16x32_f16(false, a, false, b0, (short)0, c0, false, false);
        c1 = __builtin_amdgcn_wmma_f32_16x16x32_f16(false, a, false, b1, (short)0, c1, false, false);
    }
}

template<int NK>
__device__ __forceinline__ void mma3(const _Float16* A,
                                     const _Float16* B0, const _Float16* B1,
                                     const _Float16* B2,
                                     f8& c0, f8& c1, f8& c2) {
#pragma unroll
    for (int k = 0; k < NK; ++k) {
        h16 a  = *(const h16*)(A  + k * 32);
        h16 b0 = *(const h16*)(B0 + k * 32);
        h16 b1 = *(const h16*)(B1 + k * 32);
        h16 b2 = *(const h16*)(B2 + k * 32);
        c0 = __builtin_amdgcn_wmma_f32_16x16x32_f16(false, a, false, b0, (short)0, c0, false, false);
        c1 = __builtin_amdgcn_wmma_f32_16x16x32_f16(false, a, false, b1, (short)0, c1, false, false);
        c2 = __builtin_amdgcn_wmma_f32_16x16x32_f16(false, a, false, b2, (short)0, c2, false, false);
    }
}

// ---------------- weight fp32 -> f16 conversion ----------------
extern "C" __global__ void __launch_bounds__(256)
odernn_convert(const float* __restrict__ Wih, const float* __restrict__ Whh,
               const float* __restrict__ W1,  const float* __restrict__ W2,
               const float* __restrict__ W3,  _Float16* __restrict__ ws)
{
    int i = blockIdx.x * blockDim.x + threadIdx.x;
    if (i < 768 * 64)   ws[WS_WIH + i] = (_Float16)Wih[i];
    if (i < 768 * 256)  ws[WS_WHH + i] = (_Float16)Whh[i];
    if (i < 256 * 288) {
        int r = i / 288, k = i % 288;
        ws[WS_W1P + i] = (_Float16)((k < 257) ? W1[r * 257 + k] : 0.0f);
    }
    if (i < 256 * 256) {
        ws[WS_W2 + i] = (_Float16)W2[i];
        ws[WS_W3 + i] = (_Float16)W3[i];
    }
}

// ---------------- persistent per-batch-tile ODE-RNN ----------------
extern "C" __global__ void __launch_bounds__(256)
odernn_main(const float* __restrict__ x, const float* __restrict__ t,
            const float* __restrict__ mask,
            const float* __restrict__ b_ih, const float* __restrict__ b_hh,
            const float* __restrict__ b1, const float* __restrict__ b2,
            const float* __restrict__ b3,
            const _Float16* __restrict__ ws, float* __restrict__ out)
{
    extern __shared__ __align__(128) char smem[];
    float*    sh    = (float*)(smem + SM_H);
    _Float16* sA    = (_Float16*)(smem + SM_A);
    _Float16* sB    = (_Float16*)(smem + SM_B);
    float*    sbih  = (float*)(smem + SM_BIH);
    float*    sbhh  = (float*)(smem + SM_BHH);
    float*    sb1   = (float*)(smem + SM_B1);
    float*    sb2   = (float*)(smem + SM_B2);
    float*    sb3   = (float*)(smem + SM_B3);
    float*    smask = (float*)(smem + SM_MASK);
    _Float16* sW2   = (_Float16*)(smem + SM_W2);
    _Float16* sW3   = (_Float16*)(smem + SM_W3);

    const _Float16* Wih = ws + WS_WIH;
    const _Float16* Whh = ws + WS_WHH;
    const _Float16* W1p = ws + WS_W1P;

    const int tid  = threadIdx.x;
    const int lane = tid & 31;
    const int wave = tid >> 5;          // 8 waves
    const int half = (lane >> 4) & 1;   // K-half select for A/B fragments
    const int col  = lane & 15;         // N (and A-row) index within tile
    const int b0   = blockIdx.x * BM;   // batch tile base

    // preload biases + W2/W3 into LDS, zero-init h
    for (int i = tid; i < 3 * L_SZ; i += 256) { sbih[i] = b_ih[i]; sbhh[i] = b_hh[i]; }
    for (int i = tid; i < H_SZ; i += 256)     { sb1[i] = b1[i]; sb2[i] = b2[i]; sb3[i] = b3[i]; }
    for (int i = tid; i < BM * L_SZ; i += 256) sh[i] = 0.0f;
    {
        const uint4* g2 = (const uint4*)(ws + WS_W2);
        const uint4* g3 = (const uint4*)(ws + WS_W3);
        uint4* l2 = (uint4*)sW2;
        uint4* l3 = (uint4*)sW3;
        for (int i = tid; i < (H_SZ * H_SZ) / 8; i += 256) { l2[i] = g2[i]; l3[i] = g3[i]; }
    }
    __syncthreads();

    for (int ti = 0; ti < T_SZ; ++ti) {
        // ---- stage h -> sA (f16), x[:,ti,:] -> sB (f16), mask row values ----
        for (int i = tid; i < BM * L_SZ; i += 256) {
            int m = i >> 8, n = i & 255;
            sA[m * ASTR + n] = (_Float16)sh[i];
        }
        for (int i = tid; i < BM * D_SZ; i += 256) {
            int m = i >> 6, k = i & 63;
            sB[m * ASTR + k] =
                (_Float16)x[(size_t)(b0 + m) * T_SZ * D_SZ + (size_t)ti * D_SZ + k];
        }
        if (tid < BM) smask[tid] = mask[(size_t)(b0 + tid) * T_SZ + ti];
        __syncthreads();

        // ---- fused GRU ----
#pragma unroll 1
        for (int tile = 0; tile < 2; ++tile) {
            const int n = (wave + tile * 8) * 16 + col;
            f8 ar, az, ani, anh;   // r/z fuse gi+gh into one accumulator each
#pragma unroll
            for (int r = 0; r < 8; ++r) {
                ar[r]  = sbih[n] + sbhh[n];
                az[r]  = sbih[L_SZ + n] + sbhh[L_SZ + n];
                ani[r] = sbih[2 * L_SZ + n];
                anh[r] = sbhh[2 * L_SZ + n];
            }
            const _Float16* Ax = &sB[col * ASTR + half * 16];
            const _Float16* Ah = &sA[col * ASTR + half * 16];
            mma3<2>(Ax, &Wih[(size_t)n * D_SZ + half * 16],
                        &Wih[(size_t)(L_SZ + n) * D_SZ + half * 16],
                        &Wih[(size_t)(2 * L_SZ + n) * D_SZ + half * 16],
                    ar, az, ani);
            mma3<8>(Ah, &Whh[(size_t)n * L_SZ + half * 16],
                        &Whh[(size_t)(L_SZ + n) * L_SZ + half * 16],
                        &Whh[(size_t)(2 * L_SZ + n) * L_SZ + half * 16],
                    ar, az, anh);
#pragma unroll
            for (int r = 0; r < 8; ++r) {
                const int m  = r + half * 8;
                const float hv = sh[m * L_SZ + n];
                const float rg = sigm(ar[r]);
                const float zg = sigm(az[r]);
                const float ng = tanh_fast(ani[r] + rg * anh[r]);
                const float hnew = (1.0f - zg) * ng + zg * hv;
                const float mk = smask[m];
                const float hobs = mk * hnew + (1.0f - mk) * hv;
                sh[m * L_SZ + n] = hobs;
                out[((size_t)(b0 + m) * T_SZ + ti) * L_SZ + n] = hobs;
            }
        }
        __syncthreads();

        if (ti == T_SZ - 1) break;

        const float t0 = t[ti];
        const float dt = (t[ti + 1] - t0) * (1.0f / NSUB);
        const int n0 = wave * 16 + col;
        const int n1 = (wave + 8) * 16 + col;

        // ---- RK4 neural-ODE integration ----
        for (int sub = 0; sub < NSUB; ++sub) {
            const float tt = t0 + dt * (float)sub;
            f8 acc0, acc1;

            // stage y0 = h into sA (tile owners), time + zero-pad columns
#pragma unroll
            for (int r = 0; r < 8; ++r) {
                const int m = r + half * 8;
                sA[m * ASTR + n0] = (_Float16)sh[m * L_SZ + n0];
                sA[m * ASTR + n1] = (_Float16)sh[m * L_SZ + n1];
                acc0[r] = 0.0f; acc1[r] = 0.0f;
            }
            if (tid < BM) {
                sA[tid * ASTR + 256] = (_Float16)tt;
                for (int k = 257; k < ASTR; ++k) sA[tid * ASTR + k] = (_Float16)0.0f;
            }
            __syncthreads();

            for (int stg = 0; stg < 4; ++stg) {
                // L1: tanh(sA[16x288] @ W1p.T + b1) -> sB   (W1 streamed from L2)
                {
                    f8 c0, c1;
#pragma unroll
                    for (int r = 0; r < 8; ++r) { c0[r] = sb1[n0]; c1[r] = sb1[n1]; }
                    mma2<9>(&sA[col * ASTR + half * 16],
                            &W1p[(size_t)n0 * K1P + half * 16],
                            &W1p[(size_t)n1 * K1P + half * 16], c0, c1);
#pragma unroll
                    for (int r = 0; r < 8; ++r) {
                        const int m = r + half * 8;
                        sB[m * ASTR + n0] = (_Float16)tanh_fast(c0[r]);
                        sB[m * ASTR + n1] = (_Float16)tanh_fast(c1[r]);
                    }
                }
                __syncthreads();
                // L2: tanh(sB @ W2.T + b2) -> sA   (W2 in LDS)
                {
                    f8 c0, c1;
#pragma unroll
                    for (int r = 0; r < 8; ++r) { c0[r] = sb2[n0]; c1[r] = sb2[n1]; }
                    mma2<8>(&sB[col * ASTR + half * 16],
                            &sW2[n0 * H_SZ + half * 16],
                            &sW2[n1 * H_SZ + half * 16], c0, c1);
#pragma unroll
                    for (int r = 0; r < 8; ++r) {
                        const int m = r + half * 8;
                        sA[m * ASTR + n0] = (_Float16)tanh_fast(c0[r]);
                        sA[m * ASTR + n1] = (_Float16)tanh_fast(c1[r]);
                    }
                }
                __syncthreads();
                // L3: k = sA @ W3.T + b3 (registers; W3 in LDS)
                f8 k0, k1;
                {
#pragma unroll
                    for (int r = 0; r < 8; ++r) { k0[r] = sb3[n0]; k1[r] = sb3[n1]; }
                    mma2<8>(&sA[col * ASTR + half * 16],
                            &sW3[n0 * L_SZ + half * 16],
                            &sW3[n1 * L_SZ + half * 16], k0, k1);
                }
                __syncthreads();  // all L3 reads of sA done before rewrite

                const float wgt = (stg == 1 || stg == 2) ? 2.0f : 1.0f;
                const float cs  = (stg == 2) ? dt : 0.5f * dt;
#pragma unroll
                for (int r = 0; r < 8; ++r) {
                    const int m = r + half * 8;
                    acc0[r] += wgt * k0[r];
                    acc1[r] += wgt * k1[r];
                    if (stg < 3) {
                        sA[m * ASTR + n0] = (_Float16)(sh[m * L_SZ + n0] + cs * k0[r]);
                        sA[m * ASTR + n1] = (_Float16)(sh[m * L_SZ + n1] + cs * k1[r]);
                    } else {
                        sh[m * L_SZ + n0] += (dt * (1.0f / 6.0f)) * acc0[r];
                        sh[m * L_SZ + n1] += (dt * (1.0f / 6.0f)) * acc1[r];
                    }
                }
                if (stg < 3 && tid < BM) {
                    const float ts = tt + ((stg == 2) ? dt : 0.5f * dt);
                    sA[tid * ASTR + 256] = (_Float16)ts;
                }
                __syncthreads();
            }
        }
    }
}

extern "C" void kernel_launch(void* const* d_in, const int* in_sizes, int n_in,
                              void* d_out, int out_size, void* d_ws, size_t ws_size,
                              hipStream_t stream) {
    const float* x    = (const float*)d_in[0];
    const float* t    = (const float*)d_in[1];
    const float* mask = (const float*)d_in[2];
    const float* Wih  = (const float*)d_in[3];
    const float* Whh  = (const float*)d_in[4];
    const float* bih  = (const float*)d_in[5];
    const float* bhh  = (const float*)d_in[6];
    const float* W1   = (const float*)d_in[7];
    const float* b1   = (const float*)d_in[8];
    const float* W2   = (const float*)d_in[9];
    const float* b2   = (const float*)d_in[10];
    const float* W3   = (const float*)d_in[11];
    const float* b3   = (const float*)d_in[12];
    float* out = (float*)d_out;
    _Float16* ws = (_Float16*)d_ws;

    odernn_convert<<<(768 * 256 + 255) / 256, 256, 0, stream>>>(Wih, Whh, W1, W2, W3, ws);
    odernn_main<<<B_SZ / BM, 256, SM_TOTAL, stream>>>(x, t, mask, bih, bhh, b1, b2, b3, ws, out);
}
